// RUSAwareGatingNetworkWithAttention_23648089932608
// MI455X (gfx1250) — matrix operand order repaired and checked by reading, not compiled, not run on hardware
//
#include <hip/hip_runtime.h>
#include <hip/hip_bf16.h>
#include <math.h>

// ---------------------------------------------------------------------------
// Problem constants (match reference)
// ---------------------------------------------------------------------------
constexpr int E_   = 768;
constexpr int P_   = 256;
constexpr int K_   = 32;
constexpr int V_   = 32;
constexpr int H_   = 128;
constexpr int NE_  = 8;
constexpr int B_   = 16;
constexpr int MD_  = 4;     // M_DIM
constexpr int TD_  = 512;   // T_DIM
constexpr int ROWS = B_ * MD_ * TD_;   // 32768 flattened (b,m,t) rows
constexpr int PH   = P_ + H_;          // 384 (combined width)
constexpr int G3H  = 3 * H_;           // 384 (GRU gates)
constexpr int HID  = PH / 2;           // 192

// Workspace layout (float offsets)
constexpr size_t OFF_COMBINED = 0;                                  // [ROWS, 384]
constexpr size_t OFF_QUERY    = OFF_COMBINED + (size_t)ROWS * PH;   // [ROWS, 32]
constexpr size_t OFF_GRUIN    = OFF_QUERY    + (size_t)ROWS * K_;   // [64, 512, 33]
constexpr size_t OFF_GI       = OFF_GRUIN    + (size_t)64 * TD_ * (1 + V_); // [64,512,384]
constexpr size_t OFF_HIDDEN   = OFF_GI;   // reuse: gi dead once GRU done; [ROWS,192]

typedef __bf16 v16bf  __attribute__((ext_vector_type(16)));
typedef __bf16 bf16x8 __attribute__((ext_vector_type(8)));   // 16 B
typedef __bf16 bf16x4 __attribute__((ext_vector_type(4)));   // 8 B
typedef float  v8f    __attribute__((ext_vector_type(8)));
typedef float  f4     __attribute__((ext_vector_type(4)));

// ---------------------------------------------------------------------------
// Tiled WMMA GEMM  C = act(A[M,K] @ W[K,N] + b), f32 in/out, bf16 compute.
// Block tile 128x64 (8 waves, wave32); each wave computes 32x32 via 4 WMMAs
// per 32-wide K step. LDS layouts are fragment-native:
//   As : [row][k]      row stride 40 bf16 (80 B) -> A frag = 2x ds_load_b128
//   WsT: [col][k] (transposed) same stride       -> B frag = 2x ds_load_b128
// 80 B = 20 words stride => 16 consecutive rows/cols hit distinct banks.
// Requires: Mrows % 128 == 0, Ktot % 32 == 0 (true for all call sites).
// ---------------------------------------------------------------------------
__global__ __launch_bounds__(256)
void gemm_bf16_wmma(const float* __restrict__ A, int lda,
                    const float* __restrict__ W,
                    const float* __restrict__ bias,
                    float* __restrict__ C, int ldc,
                    int Ktot, int N, int doRelu)
{
    __shared__ __align__(16) __bf16 As[128][40];   // 10240 B (cols 0..31 used)
    __shared__ __align__(16) __bf16 WsT[64][40];   //  5120 B

    const int tid  = threadIdx.x;
    const int lane = tid & 31;
    const int wave = tid >> 5;          // 0..7
    const int rt   = (wave & 3) * 32;   // wave row-tile base (4 x 32 rows)
    const int ch   = (wave >> 2) * 32;  // wave col-tile base (2 x 32 cols)
    const int lh   = lane & 15;
    const int hi   = lane >> 4;         // 0/1

    const int rowBase = blockIdx.y * 128;
    const int colBase = blockIdx.x * 64;

    v8f acc00 = {}, acc01 = {}, acc10 = {}, acc11 = {};

    for (int kc = 0; kc < Ktot; kc += 32) {
        __syncthreads();
        // ---- A tile: 128x32 f32 -> bf16. 1024 float4 chunks, 4 per thread.
        #pragma unroll
        for (int j = 0; j < 4; ++j) {
            int c  = tid + j * 256;
            int r  = c >> 3;            // 0..127
            int c4 = (c & 7) * 4;       // 0..28
            f4 v = *(const f4*)(A + (size_t)(rowBase + r) * lda + kc + c4);
            bf16x4 p;
            p[0] = (__bf16)v[0]; p[1] = (__bf16)v[1];
            p[2] = (__bf16)v[2]; p[3] = (__bf16)v[3];
            *(bf16x4*)(&As[r][c4]) = p;
        }
        // ---- W tile transposed: 64 cols x 32 k. 512 k-quad chunks, 2/thread.
        #pragma unroll
        for (int j = 0; j < 2; ++j) {
            int c   = tid + j * 256;
            int col = c & 63;
            int kq  = (c >> 6) * 4;     // 0,4,...,28
            int gc  = colBase + col;
            bf16x4 p;
            if (gc < N) {
                #pragma unroll
                for (int q = 0; q < 4; ++q)
                    p[q] = (__bf16)W[(size_t)(kc + kq + q) * N + gc];
            } else {
                p[0] = p[1] = p[2] = p[3] = (__bf16)0.0f;
            }
            *(bf16x4*)(&WsT[col][kq]) = p;
        }
        // prefetch next K slice of the A stream (global_prefetch_b8)
        if (kc + 32 < Ktot) {
            int r = tid >> 1;                   // 0..127
            __builtin_prefetch(A + (size_t)(rowBase + r) * lda + kc + 32 +
                               (tid & 1) * 16, 0, 3);
        }
        __syncthreads();

        // ---- A fragments (ISA 16-bit A layout): lane<16 -> K 0-7 & 16-23,
        //      lane>=16 -> K 8-15 & 24-31; two 16-B LDS loads each.
        v16bf a0, a1;
        {
            int r0 = rt + lh, r1 = rt + 16 + lh;
            bf16x8 lo0 = *(const bf16x8*)&As[r0][hi * 8];
            bf16x8 hi0 = *(const bf16x8*)&As[r0][hi * 8 + 16];
            bf16x8 lo1 = *(const bf16x8*)&As[r1][hi * 8];
            bf16x8 hi1 = *(const bf16x8*)&As[r1][hi * 8 + 16];
            #pragma unroll
            for (int e = 0; e < 8; ++e) {
                a0[e] = lo0[e]; a0[8 + e] = hi0[e];
                a1[e] = lo1[e]; a1[8 + e] = hi1[e];
            }
        }
        // ---- B fragments: lane%16 = column, lane<16 -> K 0-7 & 16-23,
        //      lane>=16 -> K 8-15 & 24-31; two 16-B LDS loads each.
        v16bf b0, b1;
        {
            int c0 = ch + lh, c1 = ch + 16 + lh;
            bf16x8 lo0 = *(const bf16x8*)&WsT[c0][hi * 8];
            bf16x8 hi0 = *(const bf16x8*)&WsT[c0][hi * 8 + 16];
            bf16x8 lo1 = *(const bf16x8*)&WsT[c1][hi * 8];
            bf16x8 hi1 = *(const bf16x8*)&WsT[c1][hi * 8 + 16];
            #pragma unroll
            for (int e = 0; e < 8; ++e) {
                b0[e] = lo0[e]; b0[8 + e] = hi0[e];
                b1[e] = lo1[e]; b1[8 + e] = hi1[e];
            }
        }
        acc00 = __builtin_amdgcn_wmma_f32_16x16x32_bf16(
                    false, a0, false, b0, (short)0, acc00, false, false);
        acc01 = __builtin_amdgcn_wmma_f32_16x16x32_bf16(
                    false, a0, false, b1, (short)0, acc01, false, false);
        acc10 = __builtin_amdgcn_wmma_f32_16x16x32_bf16(
                    false, a1, false, b0, (short)0, acc10, false, false);
        acc11 = __builtin_amdgcn_wmma_f32_16x16x32_bf16(
                    false, a1, false, b1, (short)0, acc11, false, false);
    }

    // ---- store: C/D layout -> VGPR r <-> M = 8*hi + r, N = lane%16
    #pragma unroll
    for (int rs = 0; rs < 2; ++rs) {
        #pragma unroll
        for (int cs = 0; cs < 2; ++cs) {
            const v8f& acc = rs ? (cs ? acc11 : acc10) : (cs ? acc01 : acc00);
            int gcol = colBase + ch + cs * 16 + lh;
            if (gcol < N) {
                float bv = bias[gcol];
                int growBase = rowBase + rt + rs * 16 + hi * 8;
                #pragma unroll
                for (int r = 0; r < 8; ++r) {
                    float v = acc[r] + bv;
                    if (doRelu) v = v > 0.0f ? v : 0.0f;
                    C[(size_t)(growBase + r) * ldc + gcol] = v;
                }
            }
        }
    }
}

// ---------------------------------------------------------------------------
// Attention: per (b,m,t) 4-way softmax over pairwise features; writes
// gru_in[n,t,0]=U, gru_in[n,t,1+v]=ctx[v].
// score_j = (r_j*(q.Wk0) + s_j*(q.Wk1) + q.bk)/sqrt(K); softmax sums to 1 so
// ctx[v] = wr*Wv0[v] + ws*Wv1[v] + bv[v] with wr=sum w_j r_j, ws=sum w_j s_j.
// ---------------------------------------------------------------------------
__global__ __launch_bounds__(256)
void attention_kernel(const float* __restrict__ query,
                      const float* __restrict__ U,
                      const float* __restrict__ R,
                      const float* __restrict__ S,
                      const float* __restrict__ Wk,   // [2,32]
                      const float* __restrict__ bk,   // [32]
                      const float* __restrict__ Wv,   // [2,32]
                      const float* __restrict__ bv,   // [32]
                      float* __restrict__ gru_in)     // [64,512,33]
{
    __shared__ float swk0[32], swk1[32], sbk[32], swv0[32], swv1[32], sbv[32];
    int tid = threadIdx.x;
    if (tid < 32) { swk0[tid] = Wk[tid]; swk1[tid] = Wk[32 + tid]; sbk[tid] = bk[tid]; }
    else if (tid < 64) { int k = tid - 32; swv0[k] = Wv[k]; swv1[k] = Wv[32 + k]; sbv[k] = bv[k]; }
    __syncthreads();

    int g = blockIdx.x * blockDim.x + tid;
    int t = g & (TD_ - 1);
    int m = (g >> 9) & (MD_ - 1);
    int b = g >> 11;

    float dA = 0.f, dB = 0.f, dC = 0.f;
    #pragma unroll
    for (int k4 = 0; k4 < 32; k4 += 4) {
        f4 q = *(const f4*)(query + (size_t)g * 32 + k4);
        #pragma unroll
        for (int q_i = 0; q_i < 4; ++q_i) {
            int k = k4 + q_i;
            dA = fmaf(q[q_i], swk0[k], dA);
            dB = fmaf(q[q_i], swk1[k], dB);
            dC = fmaf(q[q_i], sbk[k], dC);
        }
    }
    const float inv_sqrtK = 0.17677669529663687f;

    float rj[4], sj[4], sc[4];
    float mx = -1e30f;
    #pragma unroll
    for (int j = 0; j < 4; ++j) {
        size_t off = ((size_t)(b * MD_ + m) * MD_ + j) * TD_ + t;
        rj[j] = R[off];
        sj[j] = S[off];
        float s = (j == m) ? -1e9f
                           : (rj[j] * dA + sj[j] * dB + dC) * inv_sqrtK;
        sc[j] = s;
        mx = fmaxf(mx, s);
    }
    float wsum = 0.f, wj[4];
    #pragma unroll
    for (int j = 0; j < 4; ++j) { wj[j] = __expf(sc[j] - mx); wsum += wj[j]; }
    float inv = 1.0f / wsum;
    float wr = 0.f, ws = 0.f;
    #pragma unroll
    for (int j = 0; j < 4; ++j) {
        wj[j] *= inv;
        wr = fmaf(wj[j], rj[j], wr);
        ws = fmaf(wj[j], sj[j], ws);
    }

    float* dst = gru_in + (size_t)g * (1 + V_);
    dst[0] = U[g];
    #pragma unroll
    for (int v = 0; v < 32; ++v)
        dst[1 + v] = fmaf(wr, swv0[v], fmaf(ws, swv1[v], sbv[v]));
}

// ---------------------------------------------------------------------------
// gi precompute: gi[n,t,g] = b_ih[g] + sum_d gru_in[n,t,d] * W_ih[g,d]  (D=33)
// One block per (n,t): x row staged in LDS, 384 threads = one gate each.
// ---------------------------------------------------------------------------
__global__ __launch_bounds__(384)
void gi_kernel(const float* __restrict__ gru_in,  // [64,512,33]
               const float* __restrict__ W_ih,    // [384,33]
               const float* __restrict__ b_ih,    // [384]
               float* __restrict__ gi)            // [64,512,384]
{
    __shared__ float x[1 + V_];
    const int nt = blockIdx.x;            // n*512 + t
    const int gch = threadIdx.x;          // 0..383
    if (gch < 1 + V_) x[gch] = gru_in[(size_t)nt * (1 + V_) + gch];
    __syncthreads();
    const float* w = W_ih + (size_t)gch * (1 + V_);
    float acc = b_ih[gch];
    #pragma unroll
    for (int d = 0; d < 1 + V_; ++d) acc = fmaf(x[d], w[d], acc);
    gi[(size_t)nt * G3H + gch] = acc;
}

// ---------------------------------------------------------------------------
// GRU scan: one block per row n (64 blocks, 128 threads = H).
// W_hh transposed into LDS: Wt[c*384+g] = W_hh[g*128+c]  (bank-conflict-free:
// 384 % 64 == 0 so lane j always hits bank j). h kept in LDS across 512 steps.
// Writes h_t into combined[:, 256:384].
// ---------------------------------------------------------------------------
__global__ __launch_bounds__(128)
void gru_kernel(const float* __restrict__ gi,    // [64,512,384]
                const float* __restrict__ W_hh,  // [384,128]
                const float* __restrict__ b_hh,  // [384]
                float* __restrict__ combined)    // [ROWS, 384], cols 256..383
{
    __shared__ float Wt[H_ * G3H];   // 128*384 floats = 192 KB
    __shared__ float hbuf[H_];

    const int n = blockIdx.x;        // 0..63
    const int j = threadIdx.x;       // 0..127

    for (int i = j; i < H_ * G3H; i += H_) {
        int c = i / G3H, gg = i % G3H;
        Wt[i] = W_hh[(size_t)gg * H_ + c];
    }
    hbuf[j] = 0.0f;
    const float bhr = b_hh[j], bhz = b_hh[H_ + j], bhn = b_hh[2 * H_ + j];
    __syncthreads();

    for (int t = 0; t < TD_; ++t) {
        float ra = bhr, za = bhz, na = bhn;
        #pragma unroll 8
        for (int c = 0; c < H_; ++c) {
            float hc = hbuf[c];
            const float* w = Wt + (size_t)c * G3H + j;
            ra = fmaf(hc, w[0],      ra);
            za = fmaf(hc, w[H_],     za);
            na = fmaf(hc, w[2 * H_], na);
        }
        const float* g = gi + ((size_t)n * TD_ + t) * G3H;
        float ir = g[j], iz = g[H_ + j], in_ = g[2 * H_ + j];
        float r = 1.0f / (1.0f + __expf(-(ir + ra)));
        float z = 1.0f / (1.0f + __expf(-(iz + za)));
        float nn = tanhf(in_ + r * na);
        float hprev = hbuf[j];
        float hnew = (1.0f - z) * nn + z * hprev;
        __syncthreads();
        hbuf[j] = hnew;
        __syncthreads();
        combined[((size_t)n * TD_ + t) * PH + P_ + j] = hnew;
    }
}

// ---------------------------------------------------------------------------
// Logits: out[row, e] = b2[e] + sum_k hidden[row,k] * W2[k,e]   (K=192, N=8)
// ---------------------------------------------------------------------------
__global__ __launch_bounds__(256)
void logits_kernel(const float* __restrict__ hidden,  // [ROWS,192]
                   const float* __restrict__ W2,      // [192,8]
                   const float* __restrict__ b2,      // [8]
                   float* __restrict__ out)           // [ROWS,8]
{
    __shared__ float w2s[HID * NE_];
    __shared__ float b2s[NE_];
    int tid = threadIdx.x;
    for (int i = tid; i < HID * NE_; i += 256) w2s[i] = W2[i];
    if (tid < NE_) b2s[tid] = b2[tid];
    __syncthreads();

    int row = blockIdx.x * blockDim.x + tid;
    float acc[NE_];
    #pragma unroll
    for (int e = 0; e < NE_; ++e) acc[e] = b2s[e];
    const float* hrow = hidden + (size_t)row * HID;
    for (int k4 = 0; k4 < HID; k4 += 4) {
        f4 hv = *(const f4*)(hrow + k4);
        #pragma unroll
        for (int q = 0; q < 4; ++q) {
            #pragma unroll
            for (int e = 0; e < NE_; ++e)
                acc[e] = fmaf(hv[q], w2s[(k4 + q) * NE_ + e], acc[e]);
        }
    }
    #pragma unroll
    for (int e = 0; e < NE_; ++e) out[(size_t)row * NE_ + e] = acc[e];
}

// ---------------------------------------------------------------------------
// Host driver
// ---------------------------------------------------------------------------
extern "C" void kernel_launch(void* const* d_in, const int* in_sizes, int n_in,
                              void* d_out, int out_size, void* d_ws, size_t ws_size,
                              hipStream_t stream)
{
    const float* emb  = (const float*)d_in[0];
    const float* U    = (const float*)d_in[1];
    const float* R    = (const float*)d_in[2];
    const float* S    = (const float*)d_in[3];
    const float* W_tp = (const float*)d_in[4];
    const float* b_tp = (const float*)d_in[5];
    const float* W_q  = (const float*)d_in[6];
    const float* b_q  = (const float*)d_in[7];
    const float* W_k  = (const float*)d_in[8];
    const float* b_k  = (const float*)d_in[9];
    const float* W_v  = (const float*)d_in[10];
    const float* b_v  = (const float*)d_in[11];
    const float* W_ih = (const float*)d_in[12];
    const float* W_hh = (const float*)d_in[13];
    const float* b_ih = (const float*)d_in[14];
    const float* b_hh = (const float*)d_in[15];
    const float* W1   = (const float*)d_in[16];
    const float* b1   = (const float*)d_in[17];
    const float* W2   = (const float*)d_in[18];
    const float* b2   = (const float*)d_in[19];

    float* ws       = (float*)d_ws;
    float* combined = ws + OFF_COMBINED;
    float* query    = ws + OFF_QUERY;
    float* gru_in   = ws + OFF_GRUIN;
    float* gi       = ws + OFF_GI;
    float* hidden   = ws + OFF_HIDDEN;   // aliases gi (disjoint live ranges)
    float* out      = (float*)d_out;

    // 1) proc = relu(emb @ W_tp + b_tp)  -> combined[:, 0:256]
    {
        dim3 grid(P_ / 64, ROWS / 128);
        gemm_bf16_wmma<<<grid, 256, 0, stream>>>(
            emb, E_, W_tp, b_tp, combined, PH, E_, P_, 1);
    }
    // 2) query = proc @ W_q + b_q   (A = combined, K=256, lda=384, N=32)
    {
        dim3 grid(1, ROWS / 128);
        gemm_bf16_wmma<<<grid, 256, 0, stream>>>(
            combined, PH, W_q, b_q, query, K_, P_, K_, 0);
    }
    // 3) attention -> gru_in [64,512,33]
    attention_kernel<<<ROWS / 256, 256, 0, stream>>>(
        query, U, R, S, W_k, b_k, W_v, b_v, gru_in);
    // 4) gi = gru_in @ W_ih^T + b_ih   (one block per (n,t))
    gi_kernel<<<64 * TD_, 384, 0, stream>>>(gru_in, W_ih, b_ih, gi);
    // 5) GRU scan -> combined[:, 256:384]
    gru_kernel<<<64, 128, 0, stream>>>(gi, W_hh, b_hh, combined);
    // 6) hidden = relu(combined @ W1 + b1)
    {
        dim3 grid((HID + 63) / 64, ROWS / 128);
        gemm_bf16_wmma<<<grid, 256, 0, stream>>>(
            combined, PH, W1, b1, hidden, HID, PH, HID, 1);
    }
    // 7) logits = hidden @ W2 + b2 -> d_out
    logits_kernel<<<ROWS / 256, 256, 0, stream>>>(hidden, W2, b2, out);
}
